// SymptomRecommender_52682068852780
// MI455X (gfx1250) — matrix operand order repaired and checked by reading, not compiled, not run on hardware
//
#include <hip/hip_runtime.h>
#include <hip/hip_bf16.h>

// ---------------------------------------------------------------------------
// Problem constants (match reference)
// ---------------------------------------------------------------------------
#define NSYM   20000
#define NPAT   100000
#define DIM    128
#define DGDIM  16
#define HEADS1 4
#define EBASE  320000
#define BSZ    512
#define NQUERY 8
#define NCAND  512
#define ETOT   (EBASE + NSYM)   // with self loops

typedef __attribute__((ext_vector_type(16))) _Float16 v16h;
typedef __attribute__((ext_vector_type(8)))  _Float16 v8h;
typedef __attribute__((ext_vector_type(8)))  float    v8f;

__device__ __forceinline__ float wred(float v) {
    v += __shfl_xor(v, 16, 32);
    v += __shfl_xor(v, 8, 32);
    v += __shfl_xor(v, 4, 32);
    v += __shfl_xor(v, 2, 32);
    v += __shfl_xor(v, 1, 32);
    return v;
}

// ---------------------------------------------------------------------------
// WMMA engine: C[128 x N] = A(LDS f16, stride=KPAD) @ W(global f32 [KREAL,N])
// 8 waves per block, wave w owns output tiles [w*NT, (w+1)*NT), NT = N/16.
// Fully templated: loop trip counts are compile-time, EXEC statically all-ones
// around every v_wmma. When KREAL != KPAD, OOB weight rows are handled by
// CLAMPING the row index (padded A columns are exactly zero, so the products
// vanish) -- no divergent guard, no zero-fill.
// Output: raw f32 to global (TOGLOB) or f16+bias+ReLU+BN to LDS (MLP layers).
// Operand layouts: A 16x32 / B 32x16 both in contiguous k order (k = 16*(lane>>4)+half); f32 C 16x16.
// ---------------------------------------------------------------------------
template<int KPAD, int KREAL, int N, bool TOGLOB>
__device__ __forceinline__ void wmma_engine(
    const _Float16* __restrict__ sA,
    const float* __restrict__ W,
    _Float16* __restrict__ stg,                 // per-wave staging: 512 halves
    int wid, int lane,
    _Float16* __restrict__ dstLDS,
    const float* __restrict__ bias, const float* __restrict__ gamma,
    const float* __restrict__ beta, float invbn,
    float* __restrict__ dstG, long mGBase, int Mmax)
{
    constexpr int NT     = N >> 4;                 // tiles per wave
    constexpr int CNT    = (NT < 8) ? NT : 8;      // m-tiles per group
    constexpr int GROUPS = (NT + 7) / 8;           // n-groups per wave
    constexpr int KSTEPS = KPAD >> 5;

    for (int g = 0; g < GROUPS; ++g) {
        const int t  = wid * NT + g * CNT;         // scalar (wid readfirstlane'd)
        const int m0 = t & 7;
        const int n0 = (t >> 3) << 4;

        v8f acc[CNT];
#pragma unroll
        for (int i = 0; i < CNT; ++i)
#pragma unroll
            for (int j = 0; j < 8; ++j) acc[i][j] = 0.0f;

        for (int ks = 0; ks < KSTEPS; ++ks) {
            const int k0 = ks << 5;
            // ---- stage B tile (32 rows of K x 16 cols of N), swizzled ----
            int krow = k0 + lane;
            if (KREAL != KPAD) krow = (krow < KREAL) ? krow : (KREAL - 1); // clamp:
            // A cols >= KREAL are zero, so clamped rows contribute 0 exactly.
            const float4* wp = reinterpret_cast<const float4*>(W + (long)krow * N + n0);
            const float4 q0 = wp[0], q1 = wp[1], q2 = wp[2], q3 = wp[3];
            // B layout: lane' = n + 16*(k>>4), half' = k&15  (k = staging lane)
            const int lg = (lane >> 4) << 4;
            const int hf = lane & 15;
            stg[(lg + 0)  * 16 + hf] = (_Float16)q0.x;
            stg[(lg + 1)  * 16 + hf] = (_Float16)q0.y;
            stg[(lg + 2)  * 16 + hf] = (_Float16)q0.z;
            stg[(lg + 3)  * 16 + hf] = (_Float16)q0.w;
            stg[(lg + 4)  * 16 + hf] = (_Float16)q1.x;
            stg[(lg + 5)  * 16 + hf] = (_Float16)q1.y;
            stg[(lg + 6)  * 16 + hf] = (_Float16)q1.z;
            stg[(lg + 7)  * 16 + hf] = (_Float16)q1.w;
            stg[(lg + 8)  * 16 + hf] = (_Float16)q2.x;
            stg[(lg + 9)  * 16 + hf] = (_Float16)q2.y;
            stg[(lg + 10) * 16 + hf] = (_Float16)q2.z;
            stg[(lg + 11) * 16 + hf] = (_Float16)q2.w;
            stg[(lg + 12) * 16 + hf] = (_Float16)q3.x;
            stg[(lg + 13) * 16 + hf] = (_Float16)q3.y;
            stg[(lg + 14) * 16 + hf] = (_Float16)q3.z;
            stg[(lg + 15) * 16 + hf] = (_Float16)q3.w;
            asm volatile("s_wait_dscnt 0x0" ::: "memory");  // wave-local LDS flush
            const v16h bf = *reinterpret_cast<const v16h*>(stg + (lane << 4));

#pragma unroll
            for (int i = 0; i < CNT; ++i) {
                // A layout: lane = (m&15)+16*(k>>4); half = k&15 (contiguous k)
                const int mrow = ((m0 + i) << 4) + (lane & 15);
                const _Float16* ap = sA + (long)mrow * KPAD + k0 + ((lane >> 4) << 4);
                v8h lo = *reinterpret_cast<const v8h*>(ap);
                v8h hi = *reinterpret_cast<const v8h*>(ap + 8);
                v16h af;
#pragma unroll
                for (int j = 0; j < 8; ++j) { af[j] = lo[j]; af[j + 8] = hi[j]; }
                acc[i] = __builtin_amdgcn_wmma_f32_16x16x32_f16(
                    false, af, false, bf, (short)0, acc[i], false, false);
            }
        }

        // ---- store: C/D layout n = lane&15, m = vgpr + 8*(lane>=16) ----
        const int ncol = n0 + (lane & 15);
        const int mAdd = (lane >> 4) << 3;
        if (TOGLOB) {
#pragma unroll
            for (int i = 0; i < CNT; ++i) {
#pragma unroll
                for (int r = 0; r < 8; ++r) {
                    const int ml = ((m0 + i) << 4) + r + mAdd;
                    const long gm = mGBase + ml;
                    if (gm < Mmax) dstG[gm * (long)N + ncol] = acc[i][r];
                }
            }
        } else {
            const float ga = gamma[ncol];
            const float be = beta[ncol];
            const float bi = bias[ncol];
#pragma unroll
            for (int i = 0; i < CNT; ++i) {
#pragma unroll
                for (int r = 0; r < 8; ++r) {
                    const int ml = ((m0 + i) << 4) + r + mAdd;
                    float v = acc[i][r] + bi;
                    v = v > 0.f ? v : 0.f;
                    v = ga * v * invbn + be;
                    dstLDS[(long)ml * N + ncol] = (_Float16)v;
                }
            }
        }
    }
}

// ---------------------------------------------------------------------------
// Generic GEMM (GAT): C[M,N] = A[M,K] @ W[K,N], f32 in/out, f16 WMMA compute.
// ---------------------------------------------------------------------------
template<int K, int N>
__global__ __launch_bounds__(256) void k_gemm(const float* __restrict__ A,
                                              const float* __restrict__ W,
                                              float* __restrict__ C, int M)
{
    extern __shared__ char smem[];
    _Float16* sA  = reinterpret_cast<_Float16*>(smem);
    _Float16* stg = reinterpret_cast<_Float16*>(smem + (size_t)128 * K * 2);
    const int tid  = threadIdx.x;
    const int wid  = __builtin_amdgcn_readfirstlane(tid >> 5);
    const int lane = tid & 31;
    const long mBase = (long)blockIdx.x << 7;
    for (int i = tid; i < 128 * K; i += 256) {
        const int r = i / K;          // K is a power of two -> shifts
        const int c = i & (K - 1);
        const long gm = mBase + r;
        sA[i] = (_Float16)((gm < M) ? A[gm * (long)K + c] : 0.f);
    }
    __syncthreads();
    wmma_engine<K, K, N, true>(sA, W, stg + (wid << 9), wid, lane,
                               nullptr, nullptr, nullptr, nullptr, 0.f,
                               C, mBase, M);
}

// ---------------------------------------------------------------------------
// GAT small kernels
// ---------------------------------------------------------------------------
__global__ void k_demog(const float* __restrict__ gender, const float* __restrict__ age,
                        const float* __restrict__ Wd, const float* __restrict__ bd,
                        float* __restrict__ dout)
{
    const int t = blockIdx.x * 256 + threadIdx.x;
    if (t >= BSZ * DGDIM) return;
    const int b = t >> 4, j = t & 15;
    float acc = bd[j];
    acc += gender[b * 2 + 0] * Wd[0 * 16 + j];
    acc += gender[b * 2 + 1] * Wd[1 * 16 + j];
    acc += age[b * 4 + 0] * Wd[2 * 16 + j];
    acc += age[b * 4 + 1] * Wd[3 * 16 + j];
    acc += age[b * 4 + 2] * Wd[4 * 16 + j];
    acc += age[b * 4 + 3] * Wd[5 * 16 + j];
    dout[t] = acc > 0.f ? acc : 0.f;
}

__global__ void k_attn(const float* __restrict__ xp, const float* __restrict__ asrc,
                       const float* __restrict__ adst, float* __restrict__ asr,
                       float* __restrict__ ads, int Hn)
{
    __shared__ float s1[128];
    __shared__ float s2[128];
    const int n = blockIdx.x, tid = threadIdx.x;
    const int HC = Hn << 7;
    for (int h = 0; h < Hn; ++h) {
        const float v = xp[(long)n * HC + (h << 7) + tid];
        s1[tid] = v * asrc[(h << 7) + tid];
        s2[tid] = v * adst[(h << 7) + tid];
        __syncthreads();
        for (int off = 64; off > 0; off >>= 1) {
            if (tid < off) { s1[tid] += s1[tid + off]; s2[tid] += s2[tid + off]; }
            __syncthreads();
        }
        if (tid == 0) { asr[n * Hn + h] = s1[0]; ads[n * Hn + h] = s2[0]; }
        __syncthreads();
    }
}

__global__ void k_edge1(const int* __restrict__ ei, const float* __restrict__ asr,
                        const float* __restrict__ ads, float* __restrict__ we,
                        float* __restrict__ den, int E, int Nn, int Hn)
{
    const int e = blockIdx.x * 256 + threadIdx.x;
    if (e >= E + Nn) return;
    const int s = (e < E) ? ei[e] : (e - E);
    const int d = (e < E) ? ei[E + e] : (e - E);
    for (int h = 0; h < Hn; ++h) {
        float x = asr[s * Hn + h] + ads[d * Hn + h];
        x = x > 0.f ? x : 0.2f * x;                 // leaky_relu 0.2
        const float w = expf(x);                    // logits ~0 -> stable w/o max
        we[(long)e * Hn + h] = w;
        unsafeAtomicAdd(&den[d * Hn + h], w);
    }
}

__global__ void k_edge2(const int* __restrict__ ei, const float* __restrict__ xp,
                        const float* __restrict__ we, const float* __restrict__ den,
                        float* __restrict__ outb, int E, int Hn)
{
    const int e = blockIdx.x;
    const int tid = threadIdx.x;  // 128
    const int s = (e < E) ? ei[e] : (e - E);
    const int d = (e < E) ? ei[E + e] : (e - E);
    const int HC = Hn << 7;
    for (int ch = tid; ch < HC; ch += 128) {
        const int h = ch >> 7;
        const float alpha = we[(long)e * Hn + h] / den[d * Hn + h];
        unsafeAtomicAdd(&outb[(long)d * HC + ch], alpha * xp[(long)s * HC + ch]);
    }
}

__global__ void k_bias_elu(float* __restrict__ g, const float* __restrict__ bias,
                           int n, int cmask)
{
    const int i = blockIdx.x * 256 + threadIdx.x;
    if (i >= n) return;
    const float v = g[i] + bias[i & cmask];
    g[i] = v > 0.f ? v : (expf(v) - 1.f);
}

__global__ void k_bias_add(float* __restrict__ g, const float* __restrict__ bias,
                           int n, int cmask)
{
    const int i = blockIdx.x * 256 + threadIdx.x;
    if (i >= n) return;
    g[i] = g[i] + bias[i & cmask];
}

__global__ void k_norms(const float* __restrict__ se, const float* __restrict__ ge,
                        float* __restrict__ invS, float* __restrict__ invG)
{
    __shared__ float r1[128];
    __shared__ float r2[128];
    const int n = blockIdx.x, tid = threadIdx.x;
    const float a = se[(long)n * 128 + tid];
    const float g = ge[(long)n * 128 + tid];
    r1[tid] = a * a; r2[tid] = g * g;
    __syncthreads();
    for (int off = 64; off > 0; off >>= 1) {
        if (tid < off) { r1[tid] += r1[tid + off]; r2[tid] += r2[tid + off]; }
        __syncthreads();
    }
    if (tid == 0) {
        invS[n] = 1.f / fmaxf(sqrtf(r1[0]), 1e-12f);
        invG[n] = 1.f / fmaxf(sqrtf(r2[0]), 1e-12f);
    }
}

__global__ void k_qmeans(const float* __restrict__ se, const float* __restrict__ ge,
                         const int* __restrict__ qs, float* __restrict__ qns,
                         float* __restrict__ qng)
{
    __shared__ float red[128];
    const int b = blockIdx.x, tid = threadIdx.x;
    float ms = 0.f, mg = 0.f;
    for (int q = 0; q < NQUERY; ++q) {
        const int qi = qs[b * NQUERY + q];
        ms += se[(long)qi * 128 + tid];
        mg += ge[(long)qi * 128 + tid];
    }
    ms *= (1.f / NQUERY); mg *= (1.f / NQUERY);
    red[tid] = ms * ms;
    __syncthreads();
    for (int off = 64; off > 0; off >>= 1) { if (tid < off) red[tid] += red[tid + off]; __syncthreads(); }
    const float invs = 1.f / fmaxf(sqrtf(red[0]), 1e-12f);
    __syncthreads();
    qns[b * 128 + tid] = ms * invs;
    red[tid] = mg * mg;
    __syncthreads();
    for (int off = 64; off > 0; off >>= 1) { if (tid < off) red[tid] += red[tid + off]; __syncthreads(); }
    const float invg = 1.f / fmaxf(sqrtf(red[0]), 1e-12f);
    qng[b * 128 + tid] = mg * invg;
}

// ---------------------------------------------------------------------------
// Fused CF kernel: per block, 128 (patient,candidate) rows resident in LDS.
// act0 build -> per-row scalar scores -> 4 fused WMMA MLP layers -> epilogue.
// ---------------------------------------------------------------------------
struct CFArgs {
    const float *pat_emb, *sym_emb, *gat_emb, *demog, *qns, *qng, *invS, *invG;
    const int   *patient_idx, *cand;
    const float *gmfW, *gmfB, *fusionW;
    const float *W1, *W2, *W3, *W4, *W5;
    const float *b1, *b2, *b3, *b4, *b5;
    const float *g1, *g2, *g3, *g4;
    const float *e1, *e2, *e3, *e4;
    float *out;
};

__global__ __launch_bounds__(256) void k_cf(CFArgs a)
{
    extern __shared__ char smem[];
    _Float16* sA   = reinterpret_cast<_Float16*>(smem);            // 128x288 = 73728 B
    _Float16* sB   = reinterpret_cast<_Float16*>(smem + 73728);    // 128x256 = 65536 B
    _Float16* stgA = reinterpret_cast<_Float16*>(smem + 139264);   // 8x512h  =  8192 B
    int*      cid  = reinterpret_cast<int*>(smem + 147456);        //   512 B
    float*    rowS = reinterpret_cast<float*>(smem + 147968);      //  1536 B

    const int tid  = threadIdx.x;
    const int wid  = __builtin_amdgcn_readfirstlane(tid >> 5);
    const int lane = tid & 31;
    const int b = blockIdx.y;
    const int tileIdx = blockIdx.x;
    const int pi = a.patient_idx[b];
    if (tid < 128) cid[tid] = a.cand[b * NCAND + tileIdx * 128 + tid];
    __syncthreads();

    // ---- build act0: [128 x 288] f16 = concat(p, s, dg, pad) ----
    for (int i = tid; i < 128 * 288; i += 256) {
        const int r = i / 288;
        const int c = i - r * 288;
        float v;
        if (c < 128)      v = a.pat_emb[(long)pi * 128 + c];
        else if (c < 256) v = a.sym_emb[(long)cid[r] * 128 + (c - 128)];
        else if (c < 272) v = a.demog[b * 16 + (c - 256)];
        else              v = 0.f;
        sA[i] = (_Float16)v;
    }
    __syncthreads();

    // ---- per-row scalar scores (gmf, cb, gr) at f32 ----
    for (int i = 0; i < 16; ++i) {
        const int r = (wid << 4) + i;
        const int cd = cid[r];
        float sg = 0.f, sc = 0.f, sr = 0.f;
#pragma unroll
        for (int j = 0; j < 4; ++j) {
            const int c = lane + (j << 5);
            const float pv = a.pat_emb[(long)pi * 128 + c];
            const float sv = a.sym_emb[(long)cd * 128 + c];
            sg += pv * sv * a.gmfW[c];
            sc += a.qns[b * 128 + c] * sv;
            sr += a.qng[b * 128 + c] * a.gat_emb[(long)cd * 128 + c];
        }
        sg = wred(sg); sc = wred(sc); sr = wred(sr);
        if (lane == 0) {
            rowS[r * 3 + 0] = sg + a.gmfB[0];
            rowS[r * 3 + 1] = sc * a.invS[cd];
            rowS[r * 3 + 2] = sr * a.invG[cd];
        }
    }
    __syncthreads();

    const float invbn = rsqrtf(1.f + 1e-5f);
    _Float16* stg = stgA + (wid << 9);
    // L1: 288(272) -> 256
    wmma_engine<288, 272, 256, false>(sA, a.W1, stg, wid, lane, sB,
                                      a.b1, a.g1, a.e1, invbn, nullptr, 0, 0);
    __syncthreads();
    // L2: 256 -> 128
    wmma_engine<256, 256, 128, false>(sB, a.W2, stg, wid, lane, sA,
                                      a.b2, a.g2, a.e2, invbn, nullptr, 0, 0);
    __syncthreads();
    // L3: 128 -> 64
    wmma_engine<128, 128, 64, false>(sA, a.W3, stg, wid, lane, sB,
                                     a.b3, a.g3, a.e3, invbn, nullptr, 0, 0);
    __syncthreads();
    // L4: 64 -> 32
    wmma_engine<64, 64, 32, false>(sB, a.W4, stg, wid, lane, sA,
                                   a.b4, a.g4, a.e4, invbn, nullptr, 0, 0);
    __syncthreads();

    // ---- epilogue: final linear (32->1) + fusion ----
    const float w5v = a.W5[lane];
    const float f0 = a.fusionW[0], f1 = a.fusionW[1], f2 = a.fusionW[2];
    const float fm = fmaxf(f0, fmaxf(f1, f2));
    const float x0 = expf(f0 - fm), x1 = expf(f1 - fm), x2 = expf(f2 - fm);
    const float xs = x0 + x1 + x2;
    const float fw0 = x0 / xs, fw1 = x1 / xs, fw2 = x2 / xs;
    const float b5v = a.b5[0];
    for (int i = 0; i < 16; ++i) {
        const int r = (wid << 4) + i;
        const float av = (float)sA[r * 32 + lane];
        const float mlp = wred(av * w5v) + b5v;
        if (lane == 0) {
            const float cf = rowS[r * 3 + 0] + mlp;
            const float sig = 1.f / (1.f + expf(-cf));
            const float res = fw0 * sig
                            + fw1 * (rowS[r * 3 + 1] + 1.f) * 0.5f
                            + fw2 * (rowS[r * 3 + 2] + 1.f) * 0.5f;
            a.out[(long)b * NCAND + tileIdx * 128 + r] = res;
        }
    }
}

// ---------------------------------------------------------------------------
// Host launch
// ---------------------------------------------------------------------------
extern "C" void kernel_launch(void* const* d_in, const int* in_sizes, int n_in,
                              void* d_out, int out_size, void* d_ws, size_t ws_size,
                              hipStream_t stream)
{
    (void)in_sizes; (void)out_size; (void)ws_size;
    const float* sym_emb   = (const float*)d_in[0];
    const float* pat_emb   = (const float*)d_in[1];
    const float* demog_W   = (const float*)d_in[2];
    const float* demog_b   = (const float*)d_in[3];
    const float* gat1_W    = (const float*)d_in[4];
    const float* gat1_asrc = (const float*)d_in[5];
    const float* gat1_adst = (const float*)d_in[6];
    const float* gat1_b    = (const float*)d_in[7];
    const float* gat2_W    = (const float*)d_in[8];
    const float* gat2_asrc = (const float*)d_in[9];
    const float* gat2_adst = (const float*)d_in[10];
    const float* gat2_b    = (const float*)d_in[11];
    const float* gmf_W     = (const float*)d_in[12];
    const float* gmf_b     = (const float*)d_in[13];
    const float* fusion_w  = (const float*)d_in[14];

    const float *W1, *W2, *W3, *W4, *W5, *mb1, *mb2, *mb3, *mb4, *mb5;
    const float *bg1, *bg2, *bg3, *bg4, *bb1, *bb2, *bb3, *bb4;
    const int *patient_idx, *query, *edge_index, *cand;
    const float *gender, *age;
    if (n_in >= 39) {   // lists flattened into separate inputs
        W1 = (const float*)d_in[15]; W2 = (const float*)d_in[16];
        W3 = (const float*)d_in[17]; W4 = (const float*)d_in[18]; W5 = (const float*)d_in[19];
        mb1 = (const float*)d_in[20]; mb2 = (const float*)d_in[21];
        mb3 = (const float*)d_in[22]; mb4 = (const float*)d_in[23]; mb5 = (const float*)d_in[24];
        bg1 = (const float*)d_in[25]; bg2 = (const float*)d_in[26];
        bg3 = (const float*)d_in[27]; bg4 = (const float*)d_in[28];
        bb1 = (const float*)d_in[29]; bb2 = (const float*)d_in[30];
        bb3 = (const float*)d_in[31]; bb4 = (const float*)d_in[32];
        patient_idx = (const int*)d_in[33];
        gender = (const float*)d_in[34];
        age    = (const float*)d_in[35];
        query  = (const int*)d_in[36];
        edge_index = (const int*)d_in[37];
        cand   = (const int*)d_in[38];
    } else {            // lists concatenated per dict key
        const float* Wc = (const float*)d_in[15];
        W1 = Wc; W2 = Wc + 69632; W3 = Wc + 102400; W4 = Wc + 110592; W5 = Wc + 112640;
        const float* bc = (const float*)d_in[16];
        mb1 = bc; mb2 = bc + 256; mb3 = bc + 384; mb4 = bc + 448; mb5 = bc + 480;
        const float* gc = (const float*)d_in[17];
        bg1 = gc; bg2 = gc + 256; bg3 = gc + 384; bg4 = gc + 448;
        const float* ec = (const float*)d_in[18];
        bb1 = ec; bb2 = ec + 256; bb3 = ec + 384; bb4 = ec + 448;
        patient_idx = (const int*)d_in[19];
        gender = (const float*)d_in[20];
        age    = (const float*)d_in[21];
        query  = (const int*)d_in[22];
        edge_index = (const int*)d_in[23];
        cand   = (const int*)d_in[24];
    }
    float* out = (float*)d_out;

    // ---- workspace layout (floats) ----
    float* ws   = (float*)d_ws;
    float* xp1  = ws;                    // 20000*512
    float* gbuf = xp1  + 10240000;       // 20000*512  (out1 -> g in place)
    float* xp2  = gbuf + 10240000;       // 20000*128
    float* gemb = xp2  + 2560000;        // 20000*128  (out2 -> gat_emb in place)
    float* asr1 = gemb + 2560000;        // 20000*4
    float* ads1 = asr1 + 80000;          // 20000*4
    float* asr2 = ads1 + 80000;          // 20000
    float* ads2 = asr2 + 20000;          // 20000
    float* we1  = ads2 + 20000;          // 340000*4
    float* we2  = we1  + 1360000;        // 340000
    float* den1 = we2  + 340000;         // 20000*4
    float* den2 = den1 + 80000;          // 20000
    float* dmg  = den2 + 20000;          // 512*16
    float* qns  = dmg  + 8192;           // 512*128
    float* qng  = qns  + 65536;          // 512*128
    float* invS = qng  + 65536;          // 20000
    float* invG = invS + 20000;          // 20000

    // ---- zero accumulators (each call; graph-replay safe) ----
    hipMemsetAsync(gbuf, 0, (size_t)10240000 * 4, stream);
    hipMemsetAsync(gemb, 0, (size_t)2560000 * 4, stream);
    hipMemsetAsync(den1, 0, (size_t)80000 * 4, stream);
    hipMemsetAsync(den2, 0, (size_t)20000 * 4, stream);

    // ---- demographics ----
    k_demog<<<(BSZ * DGDIM + 255) / 256, 256, 0, stream>>>(gender, age, demog_W, demog_b, dmg);

    // ---- GAT layer 1 ----
    const int mBlocks = (NSYM + 127) / 128;
    k_gemm<128, 512><<<mBlocks, 256, 128 * 128 * 2 + 8192, stream>>>(sym_emb, gat1_W, xp1, NSYM);
    k_attn<<<NSYM, 128, 0, stream>>>(xp1, gat1_asrc, gat1_adst, asr1, ads1, HEADS1);
    k_edge1<<<(ETOT + 255) / 256, 256, 0, stream>>>(edge_index, asr1, ads1, we1, den1, EBASE, NSYM, HEADS1);
    k_edge2<<<ETOT, 128, 0, stream>>>(edge_index, xp1, we1, den1, gbuf, EBASE, HEADS1);
    k_bias_elu<<<(10240000 + 255) / 256, 256, 0, stream>>>(gbuf, gat1_b, 10240000, 511);

    // ---- GAT layer 2 ----
    k_gemm<512, 128><<<mBlocks, 256, 128 * 512 * 2 + 8192, stream>>>(gbuf, gat2_W, xp2, NSYM);
    k_attn<<<NSYM, 128, 0, stream>>>(xp2, gat2_asrc, gat2_adst, asr2, ads2, 1);
    k_edge1<<<(ETOT + 255) / 256, 256, 0, stream>>>(edge_index, asr2, ads2, we2, den2, EBASE, NSYM, 1);
    k_edge2<<<ETOT, 128, 0, stream>>>(edge_index, xp2, we2, den2, gemb, EBASE, 1);
    k_bias_add<<<(2560000 + 255) / 256, 256, 0, stream>>>(gemb, gat2_b, 2560000, 127);

    // ---- cosine-score prep ----
    k_norms<<<NSYM, 128, 0, stream>>>(sym_emb, gemb, invS, invG);
    k_qmeans<<<BSZ, 128, 0, stream>>>(sym_emb, gemb, query, qns, qng);

    // ---- fused CF (GMF + MLP + fusion) ----
    CFArgs a;
    a.pat_emb = pat_emb; a.sym_emb = sym_emb; a.gat_emb = gemb; a.demog = dmg;
    a.qns = qns; a.qng = qng; a.invS = invS; a.invG = invG;
    a.patient_idx = patient_idx; a.cand = cand;
    a.gmfW = gmf_W; a.gmfB = gmf_b; a.fusionW = fusion_w;
    a.W1 = W1; a.W2 = W2; a.W3 = W3; a.W4 = W4; a.W5 = W5;
    a.b1 = mb1; a.b2 = mb2; a.b3 = mb3; a.b4 = mb4; a.b5 = mb5;
    a.g1 = bg1; a.g2 = bg2; a.g3 = bg3; a.g4 = bg4;
    a.e1 = bb1; a.e2 = bb2; a.e3 = bb3; a.e4 = bb4;
    a.out = out;
    dim3 cfGrid(NCAND / 128, BSZ, 1);
    k_cf<<<cfGrid, 256, 149504, stream>>>(a);
}
